// Encoder_32203664786028
// MI455X (gfx1250) — matrix operand order repaired and checked by reading, not compiled, not run on hardware
//
#include <hip/hip_runtime.h>
#include <math.h>

// ---------------- problem constants ----------------
#define T_STEPS 1019
#define BATCH   8
#define M_ROWS  (T_STEPS * BATCH)   // 8152
#define CIN     4
#define LIN     524288
#define KERN    3072
#define KRED    (CIN * KERN)        // 12288
#define COUT    512
#define HH      512
#define H3      1536

typedef __attribute__((ext_vector_type(16))) _Float16 v16h;
typedef __attribute__((ext_vector_type(8)))  _Float16 v8h;
typedef __attribute__((ext_vector_type(8)))  float    v8f;
typedef int v4i __attribute__((vector_size(16)));

// CDNA5 async global->LDS (ASYNCcnt-tracked DMA), if the toolchain exposes it.
#if __has_builtin(__builtin_amdgcn_global_load_async_to_lds_b128)
#define HAVE_ASYNC_LDS 1
#else
#define HAVE_ASYNC_LDS 0
#endif

// Builtin signature (from hipcc diagnostics): (v4i __device__* src_global,
// v4i LDS* dst, imm int offset, imm int cpol). Casts must be exact typed
// pointers (C++ has no implicit void*->T*), non-const.
#define TO_GLB(p) ((__attribute__((address_space(1))) v4i*)(p))
#define TO_LDS(p) ((__attribute__((address_space(3))) v4i*)(p))

// Split workgroup barrier (mnemonics verified from round-1 disasm).
#define WG_SIGNAL() asm volatile("s_wait_dscnt 0x0\n\ts_barrier_signal -1" ::: "memory")
#define WG_WAIT()   asm volatile("s_barrier_wait -1" ::: "memory")

// ---------------------------------------------------------------------------
// WMMA fragment loaders (wave32, 16-bit data); this code's contiguous K order, same for A and B:
// A (16x32 MxK): lanes 0-15 row M=lane, elems 0-15 -> K0-15;
//                lanes 16-31 same rows, K16-31.
// B (32x16 KxN): stored N-major (row n, contiguous K); lane = N,
//                lanes 0-15 hold K0-15, lanes 16-31 hold K16-31.
// ---------------------------------------------------------------------------
static __device__ __forceinline__ v16h frag_a_lds(const _Float16* p0, int pitch) {
  const int lane = threadIdx.x & 31;
  const int row  = lane & 15;
  const int hi   = lane >> 4;
  const _Float16* p = p0 + row * pitch + hi * 16;
  v8h a = *(const v8h*)(p);
  v8h b = *(const v8h*)(p + 8);
  v16h r;
#pragma unroll
  for (int i = 0; i < 8; ++i) { r[i] = a[i]; r[i + 8] = b[i]; }
  return r;
}

static __device__ __forceinline__ v16h frag_b_nmajor(const _Float16* p0, int pitch) {
  const int lane = threadIdx.x & 31;
  const int col  = lane & 15;
  const int hi   = lane >> 4;
  const _Float16* p = p0 + col * pitch + hi * 16;
  v8h a = *(const v8h*)(p);
  v8h b = *(const v8h*)(p + 8);
  v16h r;
#pragma unroll
  for (int i = 0; i < 8; ++i) { r[i] = a[i]; r[i + 8] = b[i]; }
  return r;
}

// B fragment straight from a contiguous 32-byte row slice (global/L2).
static __device__ __forceinline__ v16h frag_b_rows(const _Float16* p) {
  v8h a = *(const v8h*)(p);
  v8h b = *(const v8h*)(p + 8);
  v16h r;
#pragma unroll
  for (int i = 0; i < 8; ++i) { r[i] = a[i]; r[i + 8] = b[i]; }
  return r;
}

// ---------------- f32 -> f16 bulk convert ----------------
__global__ void cvt_f32_f16(const float* __restrict__ src, _Float16* __restrict__ dst, int n) {
  int i = blockIdx.x * blockDim.x + threadIdx.x;
  int stride = gridDim.x * blockDim.x;
  for (; i < n; i += stride) dst[i] = (_Float16)src[i];
}

// bias2[n] = b_ih[n] + sum_c conv_b[c] * w_ih[n, c]   (folds conv bias into gi)
__global__ void fuse_bias(const float* __restrict__ conv_b, const float* __restrict__ w_ih,
                          const float* __restrict__ b_ih, float* __restrict__ bias2) {
  int n = blockIdx.x * blockDim.x + threadIdx.x;
  if (n >= H3) return;
  float s = b_ih[n];
  for (int c = 0; c < COUT; ++c) s += conv_b[c] * w_ih[n * COUT + c];
  bias2[n] = s;
}

// ---------------------------------------------------------------------------
// Conv1d as implicit-im2col GEMM: x16[m, co] = sum_r patch(m, r) * w(co, r)
// m = t*8 + b,  r = ci*3072 + k,  patch = input[b, ci, 512*t + k]
// Block 128 thr (4 waves). Tile M=32, N=64, K-chunk=32, double-buffered LDS,
// async global->LDS staging overlapped with WMMA when available.
// ---------------------------------------------------------------------------
__global__ __launch_bounds__(128) void conv_gemm(const _Float16* __restrict__ in16,
                                                 const _Float16* __restrict__ w16,
                                                 _Float16* __restrict__ x16) {
  __shared__ _Float16 As[2][32][40];   // pitch 40 halves = 80B (16B aligned + pad)
  __shared__ _Float16 Bs[2][64][40];
  const int tid   = threadIdx.x;
  const int wave  = tid >> 5;
  const int mtile = blockIdx.x;        // 0..254
  const int ntile = blockIdx.y;        // 0..7
  v8f acc0 = {}; v8f acc1 = {};
  const int arow = tid >> 2, aseg = tid & 3;   // A: 32 rows x (4 x 8 halves)
  const int brow = tid >> 1, bseg = tid & 1;   // B: 64 rows x (2 x 16 halves)
  int m = mtile * 32 + arow; if (m > M_ROWS - 1) m = M_ROWS - 1;
  const int t = m / BATCH, b = m % BATCH;
  const size_t abase = (size_t)b * CIN * LIN + (size_t)t * 512;
  const size_t bnrow = (size_t)(ntile * 64 + brow) * KRED;
  const int NCH = KRED / 32;           // 384

#if HAVE_ASYNC_LDS
  auto stage = [&](int kc, int buf) {
    const int ci = kc / KERN;          // chunk never straddles ci (3072 % 32 == 0)
    const int k  = (kc % KERN) + aseg * 8;
    __builtin_amdgcn_global_load_async_to_lds_b128(
        TO_GLB(in16 + abase + (size_t)ci * LIN + k),
        TO_LDS(&As[buf][arow][aseg * 8]), 0, 0);
    const _Float16* gb = w16 + bnrow + kc + bseg * 16;
    __builtin_amdgcn_global_load_async_to_lds_b128(
        TO_GLB(gb), TO_LDS(&Bs[buf][brow][bseg * 16]), 0, 0);
    __builtin_amdgcn_global_load_async_to_lds_b128(
        TO_GLB(gb + 8), TO_LDS(&Bs[buf][brow][bseg * 16 + 8]), 0, 0);
  };
  stage(0, 0);
  for (int it = 0; it < NCH; ++it) {
    const int cur = it & 1;
    if (it + 1 < NCH) {
      stage((it + 1) * 32, cur ^ 1);                     // prefetch next chunk
      asm volatile("s_wait_asynccnt 0x3" ::: "memory");  // cur chunk landed
    } else {
      asm volatile("s_wait_asynccnt 0x0" ::: "memory");
    }
    __syncthreads();
    v16h bf = frag_b_nmajor(&Bs[cur][wave * 16][0], 40);
    v16h a0 = frag_a_lds(&As[cur][0][0], 40);
    v16h a1 = frag_a_lds(&As[cur][16][0], 40);
    acc0 = __builtin_amdgcn_wmma_f32_16x16x32_f16(false, a0, false, bf, (short)0, acc0, false, false);
    acc1 = __builtin_amdgcn_wmma_f32_16x16x32_f16(false, a1, false, bf, (short)0, acc1, false, false);
    __syncthreads();                    // readers done before buffer reuse
  }
#else
  for (int kc = 0; kc < KRED; kc += 32) {
    const int ci = kc / KERN;
    const int k  = (kc % KERN) + aseg * 8;
    *(v8h*)&As[0][arow][aseg * 8] = *(const v8h*)(in16 + abase + (size_t)ci * LIN + k);
    *(v8h*)&Bs[0][brow][bseg * 16]     = *(const v8h*)(w16 + bnrow + kc + bseg * 16);
    *(v8h*)&Bs[0][brow][bseg * 16 + 8] = *(const v8h*)(w16 + bnrow + kc + bseg * 16 + 8);
    __syncthreads();
    v16h bf = frag_b_nmajor(&Bs[0][wave * 16][0], 40);
    v16h a0 = frag_a_lds(&As[0][0][0], 40);
    v16h a1 = frag_a_lds(&As[0][16][0], 40);
    acc0 = __builtin_amdgcn_wmma_f32_16x16x32_f16(false, a0, false, bf, (short)0, acc0, false, false);
    acc1 = __builtin_amdgcn_wmma_f32_16x16x32_f16(false, a1, false, bf, (short)0, acc1, false, false);
    __syncthreads();
  }
#endif
  // D layout: lanes 0-15 -> M = r, lanes 16-31 -> M = r + 8; N = lane & 15
  const int lane = tid & 31;
  const int ncol = ntile * 64 + wave * 16 + (lane & 15);
  const int mhi  = (lane >> 4) * 8;
#pragma unroll
  for (int r = 0; r < 8; ++r) {
    int m0 = mtile * 32 + r + mhi;
    int m1 = mtile * 32 + 16 + r + mhi;
    if (m0 < M_ROWS) x16[(size_t)m0 * COUT + ncol] = (_Float16)acc0[r];
    if (m1 < M_ROWS) x16[(size_t)m1 * COUT + ncol] = (_Float16)acc1[r];
  }
}

// ---------------------------------------------------------------------------
// gi[m, n] = x16[m, :] . w_ih[n, :] + bias2[n]   (M=8152, K=512, N=1536)
// ---------------------------------------------------------------------------
__global__ __launch_bounds__(128) void gi_gemm(const _Float16* __restrict__ x16,
                                               const _Float16* __restrict__ wih16,
                                               const float* __restrict__ bias2,
                                               float* __restrict__ gi) {
  __shared__ _Float16 As[2][32][40];
  __shared__ _Float16 Bs[2][64][40];
  const int tid   = threadIdx.x;
  const int wave  = tid >> 5;
  const int mtile = blockIdx.x;        // 0..254
  const int ntile = blockIdx.y;        // 0..23
  v8f acc0 = {}; v8f acc1 = {};
  const int arow = tid >> 2, aseg = tid & 3;
  const int brow = tid >> 1, bseg = tid & 1;
  int m = mtile * 32 + arow; if (m > M_ROWS - 1) m = M_ROWS - 1;
  const size_t bnrow = (size_t)(ntile * 64 + brow) * COUT;
  const int NCH = COUT / 32;           // 16

#if HAVE_ASYNC_LDS
  auto stage = [&](int kc, int buf) {
    __builtin_amdgcn_global_load_async_to_lds_b128(
        TO_GLB(x16 + (size_t)m * COUT + kc + aseg * 8),
        TO_LDS(&As[buf][arow][aseg * 8]), 0, 0);
    const _Float16* gb = wih16 + bnrow + kc + bseg * 16;
    __builtin_amdgcn_global_load_async_to_lds_b128(
        TO_GLB(gb), TO_LDS(&Bs[buf][brow][bseg * 16]), 0, 0);
    __builtin_amdgcn_global_load_async_to_lds_b128(
        TO_GLB(gb + 8), TO_LDS(&Bs[buf][brow][bseg * 16 + 8]), 0, 0);
  };
  stage(0, 0);
  for (int it = 0; it < NCH; ++it) {
    const int cur = it & 1;
    if (it + 1 < NCH) {
      stage((it + 1) * 32, cur ^ 1);
      asm volatile("s_wait_asynccnt 0x3" ::: "memory");
    } else {
      asm volatile("s_wait_asynccnt 0x0" ::: "memory");
    }
    __syncthreads();
    v16h bf = frag_b_nmajor(&Bs[cur][wave * 16][0], 40);
    v16h a0 = frag_a_lds(&As[cur][0][0], 40);
    v16h a1 = frag_a_lds(&As[cur][16][0], 40);
    acc0 = __builtin_amdgcn_wmma_f32_16x16x32_f16(false, a0, false, bf, (short)0, acc0, false, false);
    acc1 = __builtin_amdgcn_wmma_f32_16x16x32_f16(false, a1, false, bf, (short)0, acc1, false, false);
    __syncthreads();
  }
#else
  for (int kc = 0; kc < COUT; kc += 32) {
    *(v8h*)&As[0][arow][aseg * 8] = *(const v8h*)(x16 + (size_t)m * COUT + kc + aseg * 8);
    *(v8h*)&Bs[0][brow][bseg * 16]     = *(const v8h*)(wih16 + bnrow + kc + bseg * 16);
    *(v8h*)&Bs[0][brow][bseg * 16 + 8] = *(const v8h*)(wih16 + bnrow + kc + bseg * 16 + 8);
    __syncthreads();
    v16h bf = frag_b_nmajor(&Bs[0][wave * 16][0], 40);
    v16h a0 = frag_a_lds(&As[0][0][0], 40);
    v16h a1 = frag_a_lds(&As[0][16][0], 40);
    acc0 = __builtin_amdgcn_wmma_f32_16x16x32_f16(false, a0, false, bf, (short)0, acc0, false, false);
    acc1 = __builtin_amdgcn_wmma_f32_16x16x32_f16(false, a1, false, bf, (short)0, acc1, false, false);
    __syncthreads();
  }
#endif
  const int lane = tid & 31;
  const int ncol = ntile * 64 + wave * 16 + (lane & 15);
  const int mhi  = (lane >> 4) * 8;
  const float bb = bias2[ncol];
#pragma unroll
  for (int r = 0; r < 8; ++r) {
    int m0 = mtile * 32 + r + mhi;
    int m1 = mtile * 32 + 16 + r + mhi;
    if (m0 < M_ROWS) gi[(size_t)m0 * H3 + ncol] = acc0[r] + bb;
    if (m1 < M_ROWS) gi[(size_t)m1 * H3 + ncol] = acc1[r] + bb;
  }
}

// ---------------------------------------------------------------------------
// GRU scan: single persistent workgroup (1024 thr = 32 wave32), h in LDS,
// w_hh f16 streamed from L2. Per step each wave: 1 A-fragment feeds 3 WMMA
// accumulators (N-tiles wave, wave+32, wave+64). Split barrier hides gi
// global-load latency behind barrier arrival.
// ---------------------------------------------------------------------------
__global__ __launch_bounds__(1024) void gru_scan(const float* __restrict__ hidden,
                                                 const _Float16* __restrict__ whh16,
                                                 const float* __restrict__ b_hh,
                                                 const float* __restrict__ gi,
                                                 float* __restrict__ out) {
  extern __shared__ char smem[];
  _Float16* h16 = (_Float16*)smem;                 // [16][520] halves, rows 8-15 zero
  float* hf  = (float*)(smem + 16 * 520 * 2);      // [8][512]
  float* gh  = hf + BATCH * HH;                    // [8][1536]
  float* bhh = gh + BATCH * H3;                    // [1536]
  const int tid  = threadIdx.x;
  const int wave = tid >> 5;
  const int lane = tid & 31;
  const int col  = lane & 15;
  const int khi  = lane >> 4;

  for (int i = tid; i < 16 * 520; i += 1024) h16[i] = (_Float16)0.f;
  for (int i = tid; i < BATCH * HH; i += 1024) hf[i] = hidden[i];
  for (int i = tid; i < H3; i += 1024) bhh[i] = b_hh[i];
  __syncthreads();
  for (int i = tid; i < BATCH * HH; i += 1024) {
    int bb = i >> 9, j = i & 511;
    h16[bb * 520 + j] = (_Float16)hf[i];
  }
  __syncthreads();

  // hoisted w_hh fragment base pointers (constant across all 1019 steps)
  const int n0 = wave * 16, n1 = (wave + 32) * 16, n2 = (wave + 64) * 16;
  const _Float16* bp0 = whh16 + (size_t)(n0 + col) * HH + khi * 16;
  const _Float16* bp1 = whh16 + (size_t)(n1 + col) * HH + khi * 16;
  const _Float16* bp2 = whh16 + (size_t)(n2 + col) * HH + khi * 16;

  for (int t = 0; t < T_STEPS; ++t) {
    const float* gi_t = gi + (size_t)t * (BATCH * H3);
    if (t + 1 < T_STEPS)   // pull next step's gate inputs toward L2/L0
      __builtin_prefetch(gi + (size_t)(t + 1) * (BATCH * H3) + tid * 12, 0, 0);

    // phase 1: gh = [h | 0pad](16x512) @ w_hh^T ; one A-frag -> 3 WMMAs
    v8f a0 = {}, a1 = {}, a2 = {};
    for (int kc = 0; kc < HH; kc += 32) {          // 16 iterations, 48 WMMAs
      v16h af = frag_a_lds(h16 + kc, 520);
      a0 = __builtin_amdgcn_wmma_f32_16x16x32_f16(false, af, false, frag_b_rows(bp0 + kc), (short)0, a0, false, false);
      a1 = __builtin_amdgcn_wmma_f32_16x16x32_f16(false, af, false, frag_b_rows(bp1 + kc), (short)0, a1, false, false);
      a2 = __builtin_amdgcn_wmma_f32_16x16x32_f16(false, af, false, frag_b_rows(bp2 + kc), (short)0, a2, false, false);
    }
    if (lane < 16) {                                // rows 8-15 are zero padding
#pragma unroll
      for (int r = 0; r < 8; ++r) {
        gh[r * H3 + n0 + lane] = a0[r];
        gh[r * H3 + n1 + lane] = a1[r];
        gh[r * H3 + n2 + lane] = a2[r];
      }
    }

    // split barrier: signal arrival, overlap gi loads with other waves' WMMAs
    WG_SIGNAL();
    float pir[4], piz[4], pin[4];
#pragma unroll
    for (int i = 0; i < 4; ++i) {
      int idx = tid + i * 1024;
      int bb = idx >> 9, j = idx & 511;
      pir[i] = gi_t[bb * H3 + j];
      piz[i] = gi_t[bb * H3 + 512 + j];
      pin[i] = gi_t[bb * H3 + 1024 + j];
    }
    WG_WAIT();

    // phase 2: gates, h update, output store (4096 elements)
#pragma unroll
    for (int i = 0; i < 4; ++i) {
      int idx = tid + i * 1024;
      int bb = idx >> 9, j = idx & 511;
      float hr = gh[bb * H3 + j] + bhh[j];
      float hz = gh[bb * H3 + 512 + j] + bhh[512 + j];
      float hn = gh[bb * H3 + 1024 + j] + bhh[1024 + j];
      float r = 1.f / (1.f + __expf(-(pir[i] + hr)));
      float z = 1.f / (1.f + __expf(-(piz[i] + hz)));
      float n = tanhf(pin[i] + r * hn);
      float hp = hf[idx];
      float hnew = (1.f - z) * n + z * hp;
      hf[idx] = hnew;
      h16[bb * 520 + j] = (_Float16)hnew;
      out[((size_t)t * BATCH + bb) * HH + j] = hnew;
    }
    __syncthreads();
  }
  for (int i = tid; i < BATCH * HH; i += 1024)      // hT
    out[(size_t)M_ROWS * HH + i] = hf[i];
}

// ---------------------------------------------------------------------------
extern "C" void kernel_launch(void* const* d_in, const int* in_sizes, int n_in,
                              void* d_out, int out_size, void* d_ws, size_t ws_size,
                              hipStream_t stream) {
  const float* input  = (const float*)d_in[0];
  const float* hidden = (const float*)d_in[1];
  const float* conv_w = (const float*)d_in[2];
  const float* conv_b = (const float*)d_in[3];
  const float* w_ih   = (const float*)d_in[4];
  const float* w_hh   = (const float*)d_in[5];
  const float* b_ih   = (const float*)d_in[6];
  const float* b_hh   = (const float*)d_in[7];
  float* out = (float*)d_out;

  // workspace layout (~108 MB), all 16B aligned
  _Float16* in16  = (_Float16*)d_ws;                          // 16,777,216 halves
  _Float16* wc16  = in16  + (size_t)BATCH * CIN * LIN;        //  6,291,456
  _Float16* wih16 = wc16  + (size_t)COUT * KRED;              //    786,432
  _Float16* whh16 = wih16 + (size_t)H3 * COUT;                //    786,432
  _Float16* x16   = whh16 + (size_t)H3 * HH;                  //  4,173,824
  float*    gi    = (float*)(x16 + (size_t)M_ROWS * COUT);    // 12,521,472 floats
  float*    bias2 = gi + (size_t)M_ROWS * H3;                 //      1,536 floats

  cvt_f32_f16<<<2048, 256, 0, stream>>>(input,  in16,  BATCH * CIN * LIN);
  cvt_f32_f16<<<2048, 256, 0, stream>>>(conv_w, wc16,  COUT * KRED);
  cvt_f32_f16<<<512,  256, 0, stream>>>(w_ih,   wih16, H3 * COUT);
  cvt_f32_f16<<<512,  256, 0, stream>>>(w_hh,   whh16, H3 * HH);
  fuse_bias<<<6, 256, 0, stream>>>(conv_b, w_ih, b_ih, bias2);

  conv_gemm<<<dim3(255, 8),  128, 0, stream>>>(in16, wc16, x16);
  gi_gemm  <<<dim3(255, 24), 128, 0, stream>>>(x16, wih16, bias2, gi);

  const size_t smem = (size_t)16 * 520 * 2            // h16
                    + (size_t)BATCH * HH * 4          // hf
                    + (size_t)BATCH * H3 * 4          // gh
                    + (size_t)H3 * 4;                 // bhh  -> 88,320 B
  gru_scan<<<1, 1024, smem, stream>>>(hidden, whh16, b_hh, gi, out);
}
